// LiGhTPredictor_23021024706989
// MI455X (gfx1250) — matrix operand order repaired and checked by reading, not compiled, not run on hardware
//
#include <hip/hip_runtime.h>
#include <hip/hip_bf16.h>

// ---------------------------------------------------------------------------
// MI455X (gfx1250) implementation of the LiGhT predictor reference.
//  * dist_attn: pf has only 8 distinct rows -> 8x4 table once, gather per edge.
//  * GEMMs: wave32 WMMA bf16 (v_wmma_f32_16x16x32_bf16, f32 accumulate),
//    f32->bf16 convert during LDS staging.  LDS layouts chosen so every
//    fragment read is a contiguous 16B chunk (ds_load_b128) and every staging
//    store is a ds_store_b128; global_prefetch_b8 on the next K tile.
//  * Graph softmax/aggregation: float atomics (CAS-max, atomicAdd).
// ---------------------------------------------------------------------------

typedef __attribute__((ext_vector_type(16))) __bf16 v16bf;
typedef __attribute__((ext_vector_type(8)))  __bf16 v8bf;
typedef __attribute__((ext_vector_type(8)))  float  v8f;

#define BM 64
#define BN 64
#define BK 32
#define LDP 40   // LDS row stride in bf16 elems: 80B, 16B-aligned, bank-skewed

__device__ inline float geluf(float x) {
    return 0.5f * x * (1.0f + erff(x * 0.70710678118654752440f));
}

__device__ inline void atomicMaxF(float* addr, float val) {
    unsigned* ua = (unsigned*)addr;
    unsigned cur = __float_as_uint(*addr);
    while (val > __uint_as_float(cur)) {
        unsigned old = atomicCAS(ua, cur, __float_as_uint(val));
        if (old == cur) break;
        cur = old;
    }
}

// ---------------------------------------------------------------------------
// C[M,Ncol] = epi(A[M,K] @ W[K,Ncol] + bias (+ resid)), act=1 -> GELU before
// residual (matches reference).  A,W row-major f32.  M%16==0, K%32==0.
// 256 threads = 8 waves; 64x64 block tile; wave = 16 rows x two 16-col tiles.
// ---------------------------------------------------------------------------
__global__ __launch_bounds__(256)
void k_wmma_gemm(const float* __restrict__ A, int lda,
                 const float* __restrict__ W, int ldw,
                 const float* __restrict__ bias,
                 const float* __restrict__ resid, int ldr,
                 float* __restrict__ C, int ldc,
                 int M, int K, int Ncol, int act)
{
    __shared__ __align__(16) __bf16 As[BM][LDP];   // [m][k]  (k: 0..31 data)
    __shared__ __align__(16) __bf16 Bt[BN][LDP];   // [n][k]  transposed B

    const int tid  = threadIdx.x;
    const int wid  = tid >> 5;
    const int lane = tid & 31;
    const int half = lane >> 4;
    const int l16  = lane & 15;

    const int bm = blockIdx.x * BM;
    const int bn = blockIdx.y * BN;

    const int tr = wid & 3;            // tile row 0..3 (16 rows each)
    const int tc = (wid >> 2) * 2;     // tile col base: 0 or 2

    // staging assignment: one contiguous 8-elem chunk per thread per tile
    const int ar  = tid >> 2;          // A row 0..63
    const int ac  = (tid & 3) * 8;     // A k-chunk base
    const int bnn = tid & 63;          // B column n (coalesced across lanes)
    const int bkc = (tid >> 6) * 8;    // B k-chunk base

    const int gr = bm + ar;
    const int gc = bn + bnn;

    v8f acc0 = {};
    v8f acc1 = {};

    for (int k0 = 0; k0 < K; k0 += BK) {
        // ---- stage A (row chunk, 2x global_load_b128 -> ds_store_b128) ----
        v8bf av;
        if (gr < M) {
            const float* ap = A + (size_t)gr * lda + k0 + ac;
#pragma unroll
            for (int j = 0; j < 8; ++j) av[j] = (__bf16)ap[j];
        } else {
#pragma unroll
            for (int j = 0; j < 8; ++j) av[j] = (__bf16)0.0f;
        }
        *(v8bf*)&As[ar][ac] = av;

        // ---- stage B transposed (coalesced global reads per k-row) --------
        v8bf bv;
        if (gc < Ncol) {
            const float* wp = W + (size_t)(k0 + bkc) * ldw + gc;
#pragma unroll
            for (int j = 0; j < 8; ++j) bv[j] = (__bf16)wp[(size_t)j * ldw];
        } else {
#pragma unroll
            for (int j = 0; j < 8; ++j) bv[j] = (__bf16)0.0f;
        }
        *(v8bf*)&Bt[bnn][bkc] = bv;

        // ---- prefetch next K tile while this one is consumed --------------
        if (k0 + BK < K) {
            if (gr < M)
                __builtin_prefetch(A + (size_t)gr * lda + (k0 + BK) + ac, 0, 0);
            if (gc < Ncol)
                __builtin_prefetch(W + (size_t)(k0 + BK + bkc) * ldw + gc, 0, 0);
        }

        __syncthreads();

        // ---- A fragment: two contiguous 16B chunks (ds_load_b128) ---------
        v16bf af;
        {
            const int m = tr * 16 + l16;
            const v8bf* pa = (const v8bf*)&As[m][0];
            v8bf alo = pa[half];        // K = half*8 .. +7
            v8bf ahi = pa[2 + half];    // K = 16+half*8 .. +7
#pragma unroll
            for (int j = 0; j < 8; ++j) { af[j] = alo[j]; af[8 + j] = ahi[j]; }
        }
        // ---- B fragments: two 16B chunks per lane from Bt[n] --------------
        v16bf bf0, bf1;
        {
            const int n0 = tc * 16 + l16;
            const v8bf* pb0 = (const v8bf*)&Bt[n0][0];
            const v8bf* pb1 = (const v8bf*)&Bt[n0 + 16][0];
            v8bf b0a = pb0[half];           // K = half*8 .. +7
            v8bf b0b = pb0[2 + half];       // K = 16+half*8 .. +7
            v8bf b1a = pb1[half];
            v8bf b1b = pb1[2 + half];
#pragma unroll
            for (int j = 0; j < 8; ++j) {
                bf0[j] = b0a[j]; bf0[8 + j] = b0b[j];
                bf1[j] = b1a[j]; bf1[8 + j] = b1b[j];
            }
        }
        acc0 = __builtin_amdgcn_wmma_f32_16x16x32_bf16(
            false, af, false, bf0, (short)0, acc0, false, false);
        acc1 = __builtin_amdgcn_wmma_f32_16x16x32_bf16(
            false, af, false, bf1, (short)0, acc1, false, false);
        __syncthreads();
    }

    // Epilogue.  C/D layout: VGPR r holds row (r + 8*half), col = l16.
    const int mbase = bm + tr * 16 + half * 8;
    const int n0 = bn + tc * 16 + l16;
    const int n1 = n0 + 16;
#pragma unroll
    for (int r = 0; r < 8; ++r) {
        int m = mbase + r;
        if (m >= M) continue;
        if (n0 < Ncol) {
            float v = acc0[r] + (bias ? bias[n0] : 0.0f);
            if (act) v = geluf(v);
            if (resid) v += resid[(size_t)m * ldr + n0];
            C[(size_t)m * ldc + n0] = v;
        }
        if (n1 < Ncol) {
            float v = acc1[r] + (bias ? bias[n1] : 0.0f);
            if (act) v = geluf(v);
            if (resid) v += resid[(size_t)m * ldr + n1];
            C[(size_t)m * ldc + n1] = v;
        }
    }
}

// ---------------------------------------------------------------------------
// LayerNorm over D=128, one wave per row, 4 elements/lane, shfl reductions.
// ---------------------------------------------------------------------------
__global__ __launch_bounds__(256)
void k_layernorm(const float* __restrict__ X, const float* __restrict__ g,
                 const float* __restrict__ b, float* __restrict__ Y, int M)
{
    int row = blockIdx.x * (blockDim.x >> 5) + (threadIdx.x >> 5);
    if (row >= M) return;
    int lane = threadIdx.x & 31;
    const float* x = X + (size_t)row * 128;
    float v[4];
    float s = 0.0f;
#pragma unroll
    for (int i = 0; i < 4; ++i) { v[i] = x[lane + 32 * i]; s += v[i]; }
#pragma unroll
    for (int o = 16; o > 0; o >>= 1) s += __shfl_xor(s, o, 32);
    float mu = s * (1.0f / 128.0f);
    float var = 0.0f;
#pragma unroll
    for (int i = 0; i < 4; ++i) { float d = v[i] - mu; var += d * d; }
#pragma unroll
    for (int o = 16; o > 0; o >>= 1) var += __shfl_xor(var, o, 32);
    float rstd = rsqrtf(var * (1.0f / 128.0f) + 1e-5f);
    float* y = Y + (size_t)row * 128;
#pragma unroll
    for (int i = 0; i < 4; ++i) {
        int c = lane + 32 * i;
        y[c] = (v[i] - mu) * rstd * g[c] + b[c];
    }
}

// pf table: rows 0..5 = path_len_emb[plen], 6 = vp_emb, 7 = sl_emb
__global__ void k_build_tpf(const float* __restrict__ ple,
                            const float* __restrict__ vpe,
                            const float* __restrict__ sle,
                            float* __restrict__ Tpf)
{
    int t = blockIdx.x * blockDim.x + threadIdx.x;
    if (t >= 8 * 128) return;
    int r = t >> 7, c = t & 127;
    float v;
    if (r <= 5)      v = ple[r * 128 + c];
    else if (r == 6) v = vpe[c];
    else             v = sle[c];
    Tpf[t] = v;
}

// Per-edge pf row index + path length
__global__ void k_edge_idx(const int* __restrict__ path, const int* __restrict__ vp,
                           const int* __restrict__ sl, int* __restrict__ eidx,
                           int* __restrict__ eplen, int E)
{
    int e = blockIdx.x * blockDim.x + threadIdx.x;
    if (e >= E) return;
    int plen = 0;
#pragma unroll
    for (int i = 0; i < 5; ++i) {
        int pv = path[e * 5 + i];
        if (pv < -99) pv = -1;
        if (pv >= 0) plen++;
    }
    int id = plen;            // 0..5
    if (vp[e] == 1) id = 6;
    if (sl[e] == 1) id = 7;
    eidx[e]  = id;
    eplen[e] = plen;
}

// distTab[8][4] = (GELU(Tpf @ dW1 + db1)) @ dW2 + db2  (single block)
__global__ __launch_bounds__(256)
void k_dist_tab(const float* __restrict__ Tpf,
                const float* __restrict__ dW1, const float* __restrict__ db1,
                const float* __restrict__ dW2, const float* __restrict__ db2,
                float* __restrict__ distTab)
{
    __shared__ float Hd[8][128];
    int tid = threadIdx.x;
    for (int i = tid; i < 8 * 128; i += 256) {
        int r = i >> 7, c = i & 127;
        float s = db1[c];
        for (int k = 0; k < 128; ++k) s += Tpf[r * 128 + k] * dW1[k * 128 + c];
        Hd[r][c] = geluf(s);
    }
    __syncthreads();
    for (int i = tid; i < 8 * 4; i += 256) {
        int r = i >> 2, h = i & 3;
        float s = db2[h];
        for (int k = 0; k < 128; ++k) s += Hd[r][k] * dW2[k * 4 + h];
        distTab[i] = s;
    }
}

// path_h[e][d] = sum_i tab_i[path[e,i]][d] / max(plen,1)
__global__ void k_path_acc(const float* __restrict__ tabbuf,
                           const int* __restrict__ path,
                           const int* __restrict__ eplen,
                           float* __restrict__ path_h, int E, int Nn)
{
    int t = blockIdx.x * blockDim.x + threadIdx.x;
    int e = t >> 5, d = t & 31;
    if (e >= E) return;
    float s = 0.0f;
#pragma unroll
    for (int i = 0; i < 5; ++i) {
        int pv = path[e * 5 + i];
        if (pv < -99) pv = -1;
        if (pv >= 0) s += tabbuf[((size_t)i * Nn + pv) * 32 + d];
    }
    int pl = eplen[e]; if (pl < 1) pl = 1;
    path_h[(size_t)e * 32 + d] = s / (float)pl;
}

// out[M][4] = Hin[M][Kd] @ W[Kd][4] + b   (small-N projection)
__global__ void k_small_out(const float* __restrict__ Hin, const float* __restrict__ W,
                            const float* __restrict__ b, float* __restrict__ out,
                            int M, int Kd)
{
    int t = blockIdx.x * blockDim.x + threadIdx.x;
    int e = t >> 2, h = t & 3;
    if (e >= M) return;
    float s = b[h];
    for (int k = 0; k < Kd; ++k) s += Hin[(size_t)e * Kd + k] * W[k * 4 + h];
    out[t] = s;
}

// a[e][h] = scale * <q[src,h], k[dst,h]> + distTab[eidx[e]][h] + pattn[e][h]
__global__ void k_attn_logits(const float* __restrict__ qkv,
                              const int* __restrict__ src, const int* __restrict__ dst,
                              const int* __restrict__ eidx,
                              const float* __restrict__ distTab,
                              const float* __restrict__ pattn,
                              float* __restrict__ a, int E)
{
    int t = blockIdx.x * blockDim.x + threadIdx.x;
    int e = t >> 2, h = t & 3;
    if (e >= E) return;
    const float scale = 0.088388347648318447f; // 128^-0.5
    const float4* q = (const float4*)(qkv + (size_t)src[e] * 384 + h * 32);
    const float4* k = (const float4*)(qkv + (size_t)dst[e] * 384 + 128 + h * 32);
    float s = 0.0f;
#pragma unroll
    for (int d = 0; d < 8; ++d) {
        float4 qv = q[d], kv = k[d];
        s += qv.x * kv.x + qv.y * kv.y + qv.z * kv.z + qv.w * kv.w;
    }
    a[t] = s * scale + distTab[eidx[e] * 4 + h] + pattn[t];
}

__global__ void k_fill(float* __restrict__ p, float v, size_t n)
{
    size_t i = (size_t)blockIdx.x * blockDim.x + threadIdx.x;
    if (i < n) p[i] = v;
}

__global__ void k_seg_max(const float* __restrict__ a, const int* __restrict__ dst,
                          float* __restrict__ m, int E)
{
    int t = blockIdx.x * blockDim.x + threadIdx.x;
    int e = t >> 2, h = t & 3;
    if (e >= E) return;
    atomicMaxF(&m[dst[e] * 4 + h], a[t]);
}

__global__ void k_seg_expsum(float* __restrict__ a, const int* __restrict__ dst,
                             const float* __restrict__ m, float* __restrict__ s, int E)
{
    int t = blockIdx.x * blockDim.x + threadIdx.x;
    int e = t >> 2, h = t & 3;
    if (e >= E) return;
    float ex = expf(a[t] - m[dst[e] * 4 + h]);
    a[t] = ex;
    atomicAdd(&s[dst[e] * 4 + h], ex);
}

// agg[dst] += v[src] * softmax(e);  one block per edge, 128 threads
__global__ __launch_bounds__(128)
void k_aggregate(const float* __restrict__ qkv, const float* __restrict__ a,
                 const float* __restrict__ s, const int* __restrict__ src,
                 const int* __restrict__ dst, float* __restrict__ agg, int E)
{
    int e = blockIdx.x;
    if (e >= E) return;
    int d = threadIdx.x;
    int h = d >> 5;
    int de = dst[e];
    float sa = a[(size_t)e * 4 + h] / s[de * 4 + h];
    float v  = qkv[(size_t)src[e] * 384 + 256 + d];
    atomicAdd(&agg[(size_t)de * 128 + d], v * sa);
}

// ---------------------------------------------------------------------------
extern "C" void kernel_launch(void* const* d_in, const int* in_sizes, int n_in,
                              void* d_out, int out_size, void* d_ws, size_t ws_size,
                              hipStream_t stream)
{
    const float* triplet_h    = (const float*)d_in[0];
    const float* path_len_emb = (const float*)d_in[1];
    const float* vp_emb       = (const float*)d_in[2];
    const float* sl_emb       = (const float*)d_in[3];
    const float* dW1  = (const float*)d_in[4];
    const float* db1  = (const float*)d_in[5];
    const float* dW2  = (const float*)d_in[6];
    const float* db2  = (const float*)d_in[7];
    const float* tWin = (const float*)d_in[8];
    const float* tbin = (const float*)d_in[9];
    const float* tWout= (const float*)d_in[10];
    const float* tbout= (const float*)d_in[11];
    const float* pW1  = (const float*)d_in[12];
    const float* pb1  = (const float*)d_in[13];
    const float* pW2  = (const float*)d_in[14];
    const float* pb2  = (const float*)d_in[15];
    const float* ln1_g   = (const float*)d_in[16];
    const float* ln1_b   = (const float*)d_in[17];
    const float* qkv_W   = (const float*)d_in[18];
    const float* qkv_b   = (const float*)d_in[19];
    const float* res_ln_g= (const float*)d_in[20];
    const float* res_ln_b= (const float*)d_in[21];
    const float* res_W   = (const float*)d_in[22];
    const float* res_b   = (const float*)d_in[23];
    const float* ffn_Win = (const float*)d_in[24];
    const float* ffn_bin = (const float*)d_in[25];
    const float* ffn_Wout= (const float*)d_in[26];
    const float* ffn_bout= (const float*)d_in[27];
    const int* src  = (const int*)d_in[28];
    const int* dst  = (const int*)d_in[29];
    const int* path = (const int*)d_in[30];
    const int* vp   = (const int*)d_in[31];
    const int* sl   = (const int*)d_in[32];

    const int D = 128, H = 4, P = 5, DT = 32, FF = 512, L = 2;
    const int N = in_sizes[0] / D;
    const int E = in_sizes[28];

    // ---- workspace carving -------------------------------------------------
    char* ws = (char*)d_ws;
    size_t off = 0;
    auto al = [](size_t b) { return (b + 255) & ~(size_t)255; };
    auto alloc = [&](size_t bytes) -> void* {
        void* p = ws + off; off += al(bytes); return p;
    };

    float* thbuf   = (float*)alloc((size_t)N * D * 4);   // running triplet_h
    float* pattn   = (float*)alloc((size_t)E * H * 4);
    float* abuf    = (float*)alloc((size_t)E * H * 4);
    float* mbuf    = (float*)alloc((size_t)N * H * 4);
    float* sbuf    = (float*)alloc((size_t)N * H * 4);
    int*   eidx    = (int*)  alloc((size_t)E * 4);
    int*   eplen   = (int*)  alloc((size_t)E * 4);
    float* Tpf     = (float*)alloc((size_t)8 * D * 4);
    float* distTab = (float*)alloc((size_t)8 * H * 4);

    // Arena shared between the pre-layer phase and the layer phase.
    size_t p0 = al((size_t)E * DT * 4);      // pathh
    size_t p1 = al((size_t)E * DT * 4);      // phhid
    size_t p2 = al((size_t)N * DT * 4);      // tmp32
    size_t p3 = al((size_t)P * N * DT * 4);  // tabbuf
    size_t l0 = al((size_t)N * D * 4);       // hbuf
    size_t l1 = al((size_t)N * 3 * D * 4);   // qkvbuf
    size_t l2 = al((size_t)N * D * 4);       // xbuf
    size_t l3 = al((size_t)N * FF * 4);      // ffnh
    size_t l4 = al((size_t)N * D * 4);       // agg
    size_t preTot = p0 + p1 + p2 + p3;
    size_t layTot = l0 + l1 + l2 + l3 + l4;
    char* arena = (char*)alloc(preTot > layTot ? preTot : layTot);
    float* pathh  = (float*)(arena);
    float* phhid  = (float*)(arena + p0);
    float* tmp32  = (float*)(arena + p0 + p1);
    float* tabbuf = (float*)(arena + p0 + p1 + p2);
    float* hbuf   = (float*)(arena);
    float* qkvbuf = (float*)(arena + l0);
    float* xbuf   = (float*)(arena + l0 + l1);
    float* ffnh   = (float*)(arena + l0 + l1 + l2);
    float* agg    = (float*)(arena + l0 + l1 + l2 + l3);

    auto gemm = [&](const float* A, int lda, const float* W, int ldw,
                    const float* bias, const float* resid, int ldr,
                    float* Cp, int ldc, int M, int K, int Ncol, int act) {
        dim3 g((M + BM - 1) / BM, (Ncol + BN - 1) / BN);
        k_wmma_gemm<<<g, 256, 0, stream>>>(A, lda, W, ldw, bias, resid, ldr,
                                           Cp, ldc, M, K, Ncol, act);
    };

    // ---- pre-layer phase ---------------------------------------------------
    hipMemcpyAsync(thbuf, triplet_h, (size_t)N * D * 4,
                   hipMemcpyDeviceToDevice, stream);

    k_build_tpf<<<(8 * 128 + 255) / 256, 256, 0, stream>>>(path_len_emb, vp_emb,
                                                           sl_emb, Tpf);
    k_dist_tab<<<1, 256, 0, stream>>>(Tpf, dW1, db1, dW2, db2, distTab);
    k_edge_idx<<<(E + 255) / 256, 256, 0, stream>>>(path, vp, sl, eidx, eplen, E);

    // node tables tab_i = GELU(th @ tWin_i + tbin_i) @ tWout_i + tbout_i
    for (int i = 0; i < P; ++i) {
        gemm(thbuf, D, tWin + (size_t)i * D * DT, DT, tbin + i * DT,
             nullptr, 0, tmp32, DT, N, D, DT, 1);
        gemm(tmp32, DT, tWout + (size_t)i * DT * DT, DT, tbout + i * DT,
             nullptr, 0, tabbuf + (size_t)i * N * DT, DT, N, DT, DT, 0);
    }
    k_path_acc<<<((size_t)E * 32 + 255) / 256, 256, 0, stream>>>(
        tabbuf, path, eplen, pathh, E, N);
    gemm(pathh, DT, pW1, DT, pb1, nullptr, 0, phhid, DT, E, DT, DT, 1);
    k_small_out<<<((size_t)E * 4 + 255) / 256, 256, 0, stream>>>(
        phhid, pW2, pb2, pattn, E, DT);

    // ---- transformer layers ------------------------------------------------
    for (int l = 0; l < L; ++l) {
        k_layernorm<<<(N + 7) / 8, 256, 0, stream>>>(
            thbuf, ln1_g + l * D, ln1_b + l * D, hbuf, N);
        gemm(hbuf, D, qkv_W + (size_t)l * D * 3 * D, 3 * D, qkv_b + l * 3 * D,
             nullptr, 0, qkvbuf, 3 * D, N, D, 3 * D, 0);

        k_attn_logits<<<((size_t)E * 4 + 255) / 256, 256, 0, stream>>>(
            qkvbuf, src, dst, eidx, distTab, pattn, abuf, E);

        k_fill<<<((size_t)N * 4 + 255) / 256, 256, 0, stream>>>(
            mbuf, -1e30f, (size_t)N * 4);
        k_fill<<<((size_t)N * 4 + 255) / 256, 256, 0, stream>>>(
            sbuf, 0.0f, (size_t)N * 4);
        k_seg_max<<<((size_t)E * 4 + 255) / 256, 256, 0, stream>>>(
            abuf, dst, mbuf, E);
        k_seg_expsum<<<((size_t)E * 4 + 255) / 256, 256, 0, stream>>>(
            abuf, dst, mbuf, sbuf, E);

        k_fill<<<((size_t)N * 128 + 255) / 256, 256, 0, stream>>>(
            agg, 0.0f, (size_t)N * 128);
        k_aggregate<<<E, 128, 0, stream>>>(qkvbuf, abuf, sbuf, src, dst, agg, E);

        // x = th + agg @ res_W + res_b
        gemm(agg, D, res_W + (size_t)l * D * D, D, res_b + l * D,
             thbuf, D, xbuf, D, N, D, D, 0);
        // y = LN(x); ffn
        k_layernorm<<<(N + 7) / 8, 256, 0, stream>>>(
            xbuf, res_ln_g + l * D, res_ln_b + l * D, hbuf, N);
        gemm(hbuf, D, ffn_Win + (size_t)l * D * FF, FF, ffn_bin + l * FF,
             nullptr, 0, ffnh, FF, N, D, FF, 1);
        // th = x + ffnh @ ffn_Wout + ffn_bout
        gemm(ffnh, FF, ffn_Wout + (size_t)l * FF * D, D, ffn_bout + l * D,
             xbuf, D, thbuf, D, N, FF, D, 0);
    }

    hipMemcpyAsync(d_out, thbuf, (size_t)N * D * 4,
                   hipMemcpyDeviceToDevice, stream);
}
